// DecoderRNN_64269890617361
// MI455X (gfx1250) — matrix operand order repaired and checked by reading, not compiled, not run on hardware
//
#include <hip/hip_runtime.h>
#include <hip/hip_bf16.h>
#include <math.h>

// ---------------------------------------------------------------------------
// Problem constants (from reference)
// ---------------------------------------------------------------------------
#define B_   128
#define L_   31
#define T_   32
#define E_   512
#define TAG_ 512
#define K_   1024      // E + TAG == H
#define H_   1024
#define FOURH_ 4096
#define V_   30000

typedef __attribute__((ext_vector_type(16))) __bf16 bf16x16;
typedef __attribute__((ext_vector_type(8)))  float  f32x8;
typedef int v4i __attribute__((vector_size(16)));                 // 16B chunk
typedef __attribute__((address_space(1))) v4i* gptr_v4i;          // global
typedef __attribute__((address_space(3))) v4i* lptr_v4i;          // LDS

// gfx1250 async global->LDS copy path (ASYNCcnt), with portable fallback
#if __has_builtin(__builtin_amdgcn_global_load_async_to_lds_b128)
#define USE_ASYNC_LDS 1
#endif

__device__ __forceinline__ void copy16_g2l(const unsigned short* g, unsigned short* l) {
#if defined(USE_ASYNC_LDS)
    __builtin_amdgcn_global_load_async_to_lds_b128(
        (gptr_v4i)(g), (lptr_v4i)(l), 0, 0);
#else
    *(uint4*)l = *(const uint4*)g;     // global_load_b128 + ds_store_b128
#endif
}

__device__ __forceinline__ void wait_async_copies() {
#if defined(USE_ASYNC_LDS)
#if __has_builtin(__builtin_amdgcn_s_wait_asynccnt)
    __builtin_amdgcn_s_wait_asynccnt(0);
#else
    asm volatile("s_wait_asynccnt 0x0" ::: "memory");
#endif
#endif
}

__device__ __forceinline__ unsigned short f2bf(float x) {
    unsigned int u = __float_as_uint(x);
    unsigned int r = u + 0x7FFFu + ((u >> 16) & 1u);   // round-to-nearest-even
    return (unsigned short)(r >> 16);
}
__device__ __forceinline__ float sigm(float x) { return 1.0f / (1.0f + expf(-x)); }

// ---------------------------------------------------------------------------
// Small prep kernels
// ---------------------------------------------------------------------------
__global__ void convert_f32_to_bf16(const float* __restrict__ src,
                                    unsigned short* __restrict__ dst, int n) {
    for (int i = blockIdx.x * blockDim.x + threadIdx.x; i < n; i += gridDim.x * blockDim.x)
        dst[i] = f2bf(src[i]);
}

__global__ void bias_sum_kernel(const float* __restrict__ b_ih,
                                const float* __restrict__ b_hh,
                                float* __restrict__ bias) {
    int i = blockIdx.x * blockDim.x + threadIdx.x;
    if (i < FOURH_) bias[i] = b_ih[i] + b_hh[i];
}

__global__ void zero_u32_kernel(unsigned int* __restrict__ p, int n32) {
    int i = blockIdx.x * blockDim.x + threadIdx.x;
    if (i < n32) p[i] = 0u;
}

// xt[b*T + t][0:512]    = (t==0) ? features[b] : W_embed[captions[b][t-1]]
// xt[b*T + t][512:1024] = tags[b]
__global__ void build_xt_kernel(const float* __restrict__ features,
                                const float* __restrict__ tags,
                                const int*   __restrict__ captions,
                                const float* __restrict__ W_embed,
                                unsigned short* __restrict__ xt) {
    int row = blockIdx.x;                 // b*T + t
    int b = row >> 5;                     // T_ == 32
    int t = row & 31;
    const float* embrow = nullptr;
    if (t > 0) embrow = W_embed + (long)captions[b * L_ + (t - 1)] * E_;
    #pragma unroll
    for (int i = 0; i < 4; ++i) {
        int c = threadIdx.x * 4 + i;      // 0..1023 with 256 threads
        float v;
        if (c < E_) v = (t == 0) ? features[b * E_ + c] : embrow[c];
        else        v = tags[b * TAG_ + (c - E_)];
        xt[(long)row * K_ + c] = f2bf(v);
    }
}

// ---------------------------------------------------------------------------
// WMMA fragment helpers (layouts per CDNA5 ISA 7.12.2, wave32)
// A 16x32 bf16: lane(lo,hi) holds row m=lo, slots 0-7 = K[hi*8 .. hi*8+7],
//               slots 8-15 = K[16+hi*8 .. 16+hi*8+7]
// B 32x16 bf16 (B[k][n] = W[n][k]): lane(lo,hi) holds col n=lo,
//               slots 0-7 = K[hi*8 .. hi*8+7], slots 8-15 = K[16+hi*8 .. 16+hi*8+7]
// ---------------------------------------------------------------------------
__device__ __forceinline__ bf16x16 load_a_frag(const unsigned short* __restrict__ base,
                                               long row_stride, int m, int k0,
                                               int hi) {
    union { uint4 q[2]; bf16x16 v; } ua;
    const unsigned short* p = base + (long)m * row_stride + k0;
    ua.q[0] = *(const uint4*)(p + hi * 8);
    ua.q[1] = *(const uint4*)(p + 16 + hi * 8);
    return ua.v;
}
__device__ __forceinline__ bf16x16 load_b_frag(const unsigned short* __restrict__ W,
                                               int n, int k0, int hi) {
    union { uint4 q[2]; bf16x16 v; } ub;
    const unsigned short* p = W + (long)n * K_ + k0;
    ub.q[0] = *(const uint4*)(p + hi * 8);
    ub.q[1] = *(const uint4*)(p + 16 + hi * 8);
    return ub.v;
}

// ---------------------------------------------------------------------------
// Per-step gate GEMM: gates[128,4096] = [xt_t | h_prev] @ [W_ih | W_hh]^T
// grid: 128 blocks x 128 threads (512 waves); one wave -> 16 rows x 64 cols
// ---------------------------------------------------------------------------
__global__ void lstm_step_gemm(const unsigned short* __restrict__ xt_t, long xt_stride,
                               const unsigned short* __restrict__ hprev, long h_stride,
                               const unsigned short* __restrict__ Wih,
                               const unsigned short* __restrict__ Whh,
                               float* __restrict__ gates) {
    const int lane = threadIdx.x & 31;
    const int wave = threadIdx.x >> 5;
    const int w = blockIdx.x * 4 + wave;      // 0..511
    const int mTile = w >> 6;                 // 0..7
    const int nGroup = w & 63;                // 64 groups of 64 columns
    const int lo = lane & 15, hi = lane >> 4;
    const int m0 = mTile * 16;
    const int n0 = nGroup * 64;

    f32x8 acc[4] = {};
    for (int k0 = 0; k0 < 2 * K_; k0 += 32) {
        const bool lowK = (k0 < K_);
        const unsigned short* Abase = lowK ? xt_t : hprev;
        const long rs = lowK ? xt_stride : h_stride;
        const int kk = lowK ? k0 : k0 - K_;
        const unsigned short* W = lowK ? Wih : Whh;

        // prefetch next K-chunk of the weight row (global_prefetch_b8)
        __builtin_prefetch(W + (long)(n0 + lo) * K_ + kk + 32, 0, 0);

        bf16x16 a = load_a_frag(Abase, rs, m0 + lo, kk, hi);
        #pragma unroll
        for (int tN = 0; tN < 4; ++tN) {
            bf16x16 b = load_b_frag(W, n0 + tN * 16 + lo, kk, hi);
            acc[tN] = __builtin_amdgcn_wmma_f32_16x16x32_bf16(
                false, a, false, b, (short)0, acc[tN], false, false);
        }
    }
    #pragma unroll
    for (int tN = 0; tN < 4; ++tN) {
        int n = n0 + tN * 16 + lo;
        #pragma unroll
        for (int r = 0; r < 8; ++r) {
            int m = m0 + hi * 8 + r;
            gates[(long)m * FOURH_ + n] = acc[tN][r];
        }
    }
}

// ---------------------------------------------------------------------------
// LSTM cell elementwise: PyTorch gate order i,f,g,o
// ---------------------------------------------------------------------------
__global__ void lstm_cell_kernel(const float* __restrict__ gates,
                                 const float* __restrict__ bias,
                                 float* __restrict__ cbuf,
                                 unsigned short* __restrict__ hiddens,  // [B,T,H] bf16
                                 int t, int first) {
    int idx = blockIdx.x * blockDim.x + threadIdx.x;   // 0..B*H-1
    int b = idx >> 10, c = idx & (H_ - 1);
    const float* g = gates + (long)b * FOURH_;
    float gi = sigm (g[c]            + bias[c]);
    float gf = sigm (g[H_ + c]       + bias[H_ + c]);
    float gg = tanhf(g[2 * H_ + c]   + bias[2 * H_ + c]);
    float go = sigm (g[3 * H_ + c]   + bias[3 * H_ + c]);
    float cold = first ? 0.0f : cbuf[idx];
    float cn = gf * cold + gi * gg;
    float h = go * tanhf(cn);
    cbuf[idx] = cn;
    hiddens[((long)b * T_ + t) * H_ + c] = f2bf(h);
}

// ---------------------------------------------------------------------------
// Packing: off[t] = sum_{s<t} #(lengths > s); row r of packed = hiddens[b, t]
// ---------------------------------------------------------------------------
__global__ void compute_off_kernel(const int* __restrict__ lengths, int* __restrict__ off) {
    if (threadIdx.x == 0 && blockIdx.x == 0) {
        int acc = 0;
        for (int t = 0; t < T_; ++t) {
            off[t] = acc;
            int cnt = 0;
            for (int b = 0; b < B_; ++b) cnt += (lengths[b] > t) ? 1 : 0;
            acc += cnt;
        }
        off[T_] = acc;
    }
}

__global__ void pack_rows_kernel(const unsigned short* __restrict__ hiddens,
                                 const int* __restrict__ off,
                                 unsigned short* __restrict__ packed, int Nrows) {
    int r = blockIdx.x;                         // padded row index
    uint2* dst = (uint2*)(packed + (long)r * H_);
    if (r >= Nrows) { dst[threadIdx.x] = make_uint2(0u, 0u); return; }
    int t = 0;
    #pragma unroll
    for (int s = 0; s < T_; ++s) if (off[s] <= r) t = s;
    int b = r - off[t];
    const uint2* src = (const uint2*)(hiddens + ((long)b * T_ + t) * H_);
    dst[threadIdx.x] = src[threadIdx.x];        // 256 threads * 8B = 2048B = 1024 bf16
}

// ---------------------------------------------------------------------------
// Final projection: out[Nrows,30000] = packed @ W_lin^T + b_lin
// Block (256 thr / 8 waves) computes 128x80; LDS-staged, double-buffered
// async global->LDS copies (ASYNCcnt). Wave wv -> rows wv*16..+15, all 80 cols.
// LDS tiles padded to 40 elements/row to spread banks.
// ---------------------------------------------------------------------------
#define LDS_ROW 40
#define NKSTEP  (K_ / 32)

__global__ void final_gemm_kernel(const unsigned short* __restrict__ packed,
                                  const unsigned short* __restrict__ Wlin,
                                  const float* __restrict__ b_lin,
                                  float* __restrict__ out,
                                  int Nrows) {
    __shared__ unsigned short lsA[2][128 * LDS_ROW];   // 2 x 10 KB
    __shared__ unsigned short lsB[2][80  * LDS_ROW];   // 2 x 6.4 KB

    const int mBlk = blockIdx.x / 375;
    const int nBlk = blockIdx.x % 375;
    const int m0 = mBlk * 128;
    const int n0 = nBlk * 80;
    const int lane = threadIdx.x & 31;
    const int wv = threadIdx.x >> 5;           // 0..7
    const int lo = lane & 15, hi = lane >> 4;

    // ---- tile copy: A = packed[m0..m0+127][kk..kk+31], B = Wlin[n0..n0+79][kk..]
    auto stage_tile = [&](int buf, int kk) {
        #pragma unroll
        for (int i = 0; i < 2; ++i) {          // 512 A-chunks of 16B
            int ci = threadIdx.x + i * 256;
            int row = ci >> 2, ch = ci & 3;
            copy16_g2l(packed + (long)(m0 + row) * K_ + kk + ch * 8,
                       &lsA[buf][row * LDS_ROW + ch * 8]);
        }
        for (int ci = threadIdx.x; ci < 320; ci += 256) {  // 320 B-chunks
            int row = ci >> 2, ch = ci & 3;
            copy16_g2l(Wlin + (long)(n0 + row) * K_ + kk + ch * 8,
                       &lsB[buf][row * LDS_ROW + ch * 8]);
        }
    };

    f32x8 acc[5] = {};
    stage_tile(0, 0);                          // prologue
    for (int ks = 0; ks < NKSTEP; ++ks) {
        const int cur = ks & 1;
        wait_async_copies();                   // tile ks landed in LDS
        __syncthreads();                       // all waves done with prior buffer
        if (ks + 1 < NKSTEP) stage_tile(cur ^ 1, (ks + 1) * 32);

        // A fragment from LDS (ds_load_b128 x2)
        union { uint4 q[2]; bf16x16 v; } ua;
        const unsigned short* pa = &lsA[cur][(wv * 16 + lo) * LDS_ROW];
        ua.q[0] = *(const uint4*)(pa + hi * 8);
        ua.q[1] = *(const uint4*)(pa + 16 + hi * 8);
        #pragma unroll
        for (int tN = 0; tN < 5; ++tN) {
            union { uint4 q[2]; bf16x16 v; } ub;
            const unsigned short* pb = &lsB[cur][(tN * 16 + lo) * LDS_ROW];
            ub.q[0] = *(const uint4*)(pb + hi * 8);
            ub.q[1] = *(const uint4*)(pb + 16 + hi * 8);
            acc[tN] = __builtin_amdgcn_wmma_f32_16x16x32_bf16(
                false, ua.v, false, ub.v, (short)0, acc[tN], false, false);
        }
    }

    #pragma unroll
    for (int tN = 0; tN < 5; ++tN) {
        int n = n0 + tN * 16 + lo;
        float bl = b_lin[n];
        #pragma unroll
        for (int r = 0; r < 8; ++r) {
            int gm = m0 + wv * 16 + hi * 8 + r;
            if (gm < Nrows) out[(long)gm * V_ + n] = acc[tN][r] + bl;
        }
    }
}

// ---------------------------------------------------------------------------
// Host launcher
// ---------------------------------------------------------------------------
extern "C" void kernel_launch(void* const* d_in, const int* in_sizes, int n_in,
                              void* d_out, int out_size, void* d_ws, size_t ws_size,
                              hipStream_t stream) {
    const float* features = (const float*)d_in[0];
    const float* tags     = (const float*)d_in[1];
    const int*   captions = (const int*)  d_in[2];
    const int*   lengths  = (const int*)  d_in[3];
    const float* W_embed  = (const float*)d_in[4];
    const float* W_ih     = (const float*)d_in[5];
    const float* W_hh     = (const float*)d_in[6];
    const float* b_ih     = (const float*)d_in[7];
    const float* b_hh     = (const float*)d_in[8];
    const float* W_lin    = (const float*)d_in[9];
    const float* b_lin    = (const float*)d_in[10];
    float* out = (float*)d_out;

    // ---- workspace carve-up (256B aligned) ----
    char* ws = (char*)d_ws;
    size_t pos = 0;
    auto carve = [&](size_t bytes) { char* p = ws + pos; pos += (bytes + 255) & ~(size_t)255; return p; };
    unsigned short* wih_bf  = (unsigned short*)carve((size_t)FOURH_ * K_ * 2);   // 8 MB
    unsigned short* whh_bf  = (unsigned short*)carve((size_t)FOURH_ * K_ * 2);   // 8 MB
    unsigned short* wlin_bf = (unsigned short*)carve((size_t)V_ * K_ * 2);       // 61.4 MB
    unsigned short* xt      = (unsigned short*)carve((size_t)B_ * T_ * K_ * 2);  // 8 MB
    unsigned short* hiddens = (unsigned short*)carve((size_t)B_ * T_ * H_ * 2);  // 8 MB
    unsigned short* packed  = (unsigned short*)carve((size_t)B_ * T_ * H_ * 2);  // 8 MB (padded rows)
    float*          gates   = (float*)carve((size_t)B_ * FOURH_ * 4);            // 2 MB
    float*          cbuf    = (float*)carve((size_t)B_ * H_ * 4);                // 0.5 MB
    float*          bias    = (float*)carve((size_t)FOURH_ * 4);
    int*            offarr  = (int*)carve((size_t)(T_ + 1) * 4);
    unsigned short* hzero   = (unsigned short*)carve((size_t)H_ * 2);

    const int Nrows = out_size / V_;            // sum(lengths), known from out_size
    const int Mblk  = (Nrows + 127) / 128;      // padded M blocks (Mblk*128 <= 4096)

    // ---- prep ----
    convert_f32_to_bf16<<<1024, 256, 0, stream>>>(W_ih,  wih_bf,  FOURH_ * K_);
    convert_f32_to_bf16<<<1024, 256, 0, stream>>>(W_hh,  whh_bf,  FOURH_ * K_);
    convert_f32_to_bf16<<<2048, 256, 0, stream>>>(W_lin, wlin_bf, V_ * K_);
    bias_sum_kernel<<<FOURH_ / 256, 256, 0, stream>>>(b_ih, b_hh, bias);
    build_xt_kernel<<<B_ * T_, 256, 0, stream>>>(features, tags, captions, W_embed, xt);
    zero_u32_kernel<<<2, 256, 0, stream>>>((unsigned int*)hzero, H_ / 2);

    // ---- LSTM scan (sequential over T) ----
    for (int t = 0; t < T_; ++t) {
        const unsigned short* hprev = (t == 0) ? hzero : (hiddens + (long)(t - 1) * H_);
        const long hs = (t == 0) ? 0L : (long)T_ * H_;
        lstm_step_gemm<<<128, 128, 0, stream>>>(xt + (long)t * K_, (long)T_ * K_,
                                                hprev, hs, wih_bf, whh_bf, gates);
        lstm_cell_kernel<<<(B_ * H_) / 256, 256, 0, stream>>>(gates, bias, cbuf,
                                                              hiddens, t, t == 0 ? 1 : 0);
    }

    // ---- pack + final projection ----
    compute_off_kernel<<<1, 32, 0, stream>>>(lengths, offarr);
    pack_rows_kernel<<<Mblk * 128, 256, 0, stream>>>(hiddens, offarr, packed, Nrows);
    final_gemm_kernel<<<Mblk * 375, 256, 0, stream>>>(packed, wlin_bf, b_lin, out, Nrows);
}
